// VQVAE_3556232921602
// MI455X (gfx1250) — matrix operand order repaired and checked by reading, not compiled, not run on hardware
//
#include <hip/hip_runtime.h>
#include <hip/hip_bf16.h>
#include <math.h>

// CDNA5 WMMA fragment types (wave32): 16 halves = 8 VGPRs, 8 floats = 8 VGPRs.
typedef __attribute__((ext_vector_type(16))) _Float16 v16h;
typedef __attribute__((ext_vector_type(8)))  float    v8f;

#define EPSBN 1e-5f

// Generic pointers into LDS are {SHARED_BASE[63:32], lds_offset[31:0]}
// (ISA 10.2 aperture mapping): low 32 bits are the DS byte address.
__device__ __forceinline__ unsigned lds_off32(const void* p) {
    return (unsigned)(unsigned long long)p;
}

// ---------------------------------------------------------------------------
// conv1: x[128,1,256,256] -> y1 raw [128,16,128,128], k4 s2 p1.
// Wave computes 16(outch) x 16(pixels); K=16 taps padded to 32.
// Epilogue: deterministic per-block per-channel sum/sumsq partials.
// part[blk*32 + c]     = sum   (c in 0..15)
// part[blk*32 + 16 + c] = sumsq
// ---------------------------------------------------------------------------
__global__ void conv1_wmma_kernel(const float* __restrict__ x,
                                  const float* __restrict__ W1,
                                  const float* __restrict__ b1,
                                  float* __restrict__ y1,
                                  float* __restrict__ part) {
    __shared__ float wred[8][32];
    const int lane = threadIdx.x & 31;
    const int wave = threadIdx.x >> 5;
    const int tile = blockIdx.x * 8 + wave;   // 131072 tiles: b(128)*oy(128)*oxg(8)
    const int b    = tile >> 10;
    const int rem  = tile & 1023;
    const int oy   = rem >> 3;
    const int ox0  = (rem & 7) << 4;

    const int m   = lane & 15;                // output channel (A row)
    const int kbA = (lane < 16) ? 0 : 8;
    const int kbB = (lane < 16) ? 0 : 16;
    const int ox  = ox0 + (lane & 15);

    v16h a, bm;
    #pragma unroll
    for (int e = 0; e < 16; ++e) {            // A: lane<16 K=0..15, lane>=16 K=16..31
        int k = kbB + e;
        a[e] = (k < 16) ? (_Float16)W1[m * 16 + k] : (_Float16)0.f;
    }
    #pragma unroll
    for (int e = 0; e < 16; ++e) {            // B: lane<16 K=0..15, lane>=16 K=16..31
        int k = kbB + e;
        float v = 0.f;
        if (k < 16) {
            int r = k >> 2, s = k & 3;
            int iy = 2 * oy - 1 + r;
            int ix = 2 * ox - 1 + s;
            if (iy >= 0 && iy < 256 && ix >= 0 && ix < 256)
                v = x[(size_t)b * 65536 + iy * 256 + ix];
        }
        bm[e] = (_Float16)v;
    }
    v8f c = {};
    c = __builtin_amdgcn_wmma_f32_16x16x32_f16(false, a, false, bm, (short)0, c, false, false);

    const int mb = (lane < 16) ? 0 : 8;
    float vals[8];
    #pragma unroll
    for (int r = 0; r < 8; ++r) {
        int mm = mb + r;
        float v = c[r] + b1[mm];
        vals[r] = v;
        y1[(((size_t)b * 16 + mm) * 128 + oy) * 128 + ox] = v;
    }

    // ---- BN statistics epilogue (deterministic trees only) ----
    float sums[8], sqs[8];
    #pragma unroll
    for (int r = 0; r < 8; ++r) { sums[r] = vals[r]; sqs[r] = vals[r] * vals[r]; }
    #pragma unroll
    for (int mask = 1; mask < 16; mask <<= 1) {
        #pragma unroll
        for (int r = 0; r < 8; ++r) {
            sums[r] += __shfl_xor(sums[r], mask, 32);
            sqs[r]  += __shfl_xor(sqs[r],  mask, 32);
        }
    }
    if ((lane & 15) == 0) {
        #pragma unroll
        for (int r = 0; r < 8; ++r) {
            wred[wave][mb + r]      = sums[r];
            wred[wave][16 + mb + r] = sqs[r];
        }
    }
    __syncthreads();
    if (threadIdx.x < 32) {
        float t = 0.f;
        #pragma unroll
        for (int w = 0; w < 8; ++w) t += wred[w][threadIdx.x];
        part[(size_t)blockIdx.x * 32 + threadIdx.x] = t;
    }
}

// ---------------------------------------------------------------------------
// Per-channel: reduce block partials, fold batch stats + gamma/beta into
// y_bn = y*scale[c] + shift[c].  Grid = C blocks, 256 threads.
// part layout: part[i*2C + c] = sum, part[i*2C + C + c] = sumsq.
// ---------------------------------------------------------------------------
__global__ void stats_scale_kernel(const float* __restrict__ part, int nblk, int C,
                                   const float* __restrict__ g, const float* __restrict__ be,
                                   float invcnt,
                                   float* __restrict__ scale, float* __restrict__ shift) {
    __shared__ float ss[256], ss2[256];
    const int c = blockIdx.x;
    float s = 0.f, s2 = 0.f;
    for (int i = threadIdx.x; i < nblk; i += 256) {
        s  += part[(size_t)i * 2 * C + c];
        s2 += part[(size_t)i * 2 * C + C + c];
    }
    ss[threadIdx.x] = s; ss2[threadIdx.x] = s2;
    __syncthreads();
    for (int off = 128; off > 0; off >>= 1) {
        if ((int)threadIdx.x < off) {
            ss[threadIdx.x]  += ss[threadIdx.x + off];
            ss2[threadIdx.x] += ss2[threadIdx.x + off];
        }
        __syncthreads();
    }
    if (threadIdx.x == 0) {
        float mean = ss[0] * invcnt;
        float var  = ss2[0] * invcnt - mean * mean;
        float sc   = g[c] * rsqrtf(var + EPSBN);
        scale[c] = sc;
        shift[c] = be[c] - mean * sc;
    }
}

// ---------------------------------------------------------------------------
// conv2: y1 raw -> y2 raw [128,4,64,64], k4 s2 p1, K=256 (8 WMMAs/tile).
// Block computes a 2-row x 64-col output slab; its 16ch x 6row x 128col input
// patch (48KB) is staged into LDS with global_load_async_to_lds_b128
// (ASYNCcnt), then BN(layer1)+ReLU is applied in LDS (padding stays zero).
// Epilogue: per-block stats partials, part[blk*8 + {c | 4+c}].
// ---------------------------------------------------------------------------
__global__ void conv2_wmma_kernel(const float* __restrict__ y1,
                                  const float* __restrict__ scale1,
                                  const float* __restrict__ shift1,
                                  const float* __restrict__ W2,
                                  const float* __restrict__ b2,
                                  float* __restrict__ y2,
                                  float* __restrict__ part) {
    __shared__ __align__(16) float patch[16 * 6 * 128];   // 48KB of the 320KB WGP LDS
    __shared__ float wred[8][8];

    const int lane = threadIdx.x & 31;
    const int wave = threadIdx.x >> 5;
    const int b    = blockIdx.x >> 5;       // 4096 blocks: b(128) * oyp(32)
    const int oyp  = blockIdx.x & 31;
    const int iyBase = 4 * oyp - 1;

    // ---- stage input patch via async DMA-style copies (16B per lane/op) ----
    const float* src = y1 + (size_t)b * 16 * 128 * 128;
    #pragma unroll
    for (int t = 0; t < 12; ++t) {
        int e4  = t * 256 + threadIdx.x;    // 3072 float4 tiles
        int cin = e4 / 192;                 // 6 rows * 32 float4 per channel
        int rem = e4 - cin * 192;
        int rr  = rem >> 5;
        int c4  = rem & 31;
        int iy  = iyBase + rr;
        float* dst = &patch[(cin * 6 + rr) * 128 + c4 * 4];
        if (iy >= 0 && iy < 128) {
            const float* gsrc = src + ((size_t)cin * 128 + iy) * 128 + c4 * 4;
            unsigned lo = lds_off32(dst);
            asm volatile("global_load_async_to_lds_b128 %0, %1, off"
                         :: "v"(lo), "v"(gsrc) : "memory");
        } else {
            dst[0] = 0.f; dst[1] = 0.f; dst[2] = 0.f; dst[3] = 0.f;
        }
    }
    asm volatile("s_wait_asynccnt 0x0" ::: "memory");
    __syncthreads();

    // ---- BN(layer1)+ReLU in LDS; conv2's zero padding must stay zero ----
    #pragma unroll
    for (int t = 0; t < 48; ++t) {
        int e   = t * 256 + threadIdx.x;    // 12288 floats
        int cin = e / 768;
        int rr  = (e / 128) % 6;
        int iy  = iyBase + rr;
        if (iy >= 0 && iy < 128) {
            float v = patch[e] * scale1[cin] + shift1[cin];
            patch[e] = v > 0.f ? v : 0.f;
        }
    }
    __syncthreads();

    // ---- WMMA: M=4 outch (rows 4..15 zero), N=16 pixels, K=256 ----
    const int oyw = wave >> 2;              // 0/1: which output row of the pair
    const int oy  = 2 * oyp + oyw;
    const int ox0 = (wave & 3) << 4;
    const int m   = lane & 15;
    const int kbA = (lane < 16) ? 0 : 8;
    const int kbB = (lane < 16) ? 0 : 16;
    const int ox  = ox0 + (lane & 15);

    v8f c = {};
    #pragma unroll
    for (int kc = 0; kc < 8; ++kc) {
        v16h a, bm;
        #pragma unroll
        for (int e = 0; e < 16; ++e) {
            int kg = kc * 32 + kbB + e; // kg = cin*16+r*4+s
            a[e] = (m < 4) ? (_Float16)W2[m * 256 + kg] : (_Float16)0.f;
        }
        #pragma unroll
        for (int e = 0; e < 16; ++e) {
            int kg  = kc * 32 + kbB + e;
            int cin = kg >> 4;
            int r   = (kg >> 2) & 3, s = kg & 3;
            int rr  = 2 * oyw + r;          // row within staged patch
            int ix  = 2 * ox - 1 + s;
            float v = 0.f;
            if (ix >= 0 && ix < 128) v = patch[(cin * 6 + rr) * 128 + ix];
            bm[e] = (_Float16)v;
        }
        c = __builtin_amdgcn_wmma_f32_16x16x32_f16(false, a, false, bm, (short)0, c, false, false);
    }

    float vals[4] = {0.f, 0.f, 0.f, 0.f};
    if (lane < 16) {
        #pragma unroll
        for (int r = 0; r < 4; ++r) {
            vals[r] = c[r] + b2[r];
            y2[(((size_t)b * 4 + r) * 64 + oy) * 64 + ox] = vals[r];
        }
    }
    // ---- stats epilogue (lanes >= 16 contribute zeros) ----
    float sums[4], sqs[4];
    #pragma unroll
    for (int r = 0; r < 4; ++r) { sums[r] = vals[r]; sqs[r] = vals[r] * vals[r]; }
    #pragma unroll
    for (int mask = 1; mask < 16; mask <<= 1) {
        #pragma unroll
        for (int r = 0; r < 4; ++r) {
            sums[r] += __shfl_xor(sums[r], mask, 32);
            sqs[r]  += __shfl_xor(sqs[r],  mask, 32);
        }
    }
    if (lane == 0) {
        #pragma unroll
        for (int r = 0; r < 4; ++r) { wred[wave][r] = sums[r]; wred[wave][4 + r] = sqs[r]; }
    }
    __syncthreads();
    if (threadIdx.x < 8) {
        float t = 0.f;
        #pragma unroll
        for (int w = 0; w < 8; ++w) t += wred[w][threadIdx.x];
        part[(size_t)blockIdx.x * 8 + threadIdx.x] = t;
    }
}

// ---------------------------------------------------------------------------
// conv3 (1x1, 4->2) + VQ + per-block loss partial. BN(layer2)+ReLU applied
// to y2 on load. Grid = 2048 x 256 == 128*64*64 threads.
// ---------------------------------------------------------------------------
__global__ void conv3_vq_kernel(const float* __restrict__ y2,
                                const float* __restrict__ scale2,
                                const float* __restrict__ shift2,
                                const float* __restrict__ W3, const float* __restrict__ b3,
                                const float* __restrict__ emb,
                                float* __restrict__ q, float* __restrict__ lossPart) {
    __shared__ float ls[256];
    const int p   = blockIdx.x * blockDim.x + threadIdx.x;
    const int b   = p >> 12;
    const int pix = p & 4095;

    float in[4];
    #pragma unroll
    for (int cin = 0; cin < 4; ++cin) {
        float v = y2[((size_t)b * 4 + cin) * 4096 + pix] * scale2[cin] + shift2[cin];
        in[cin] = v > 0.f ? v : 0.f;
    }
    float z0 = b3[0] + W3[0] * in[0] + W3[1] * in[1] + W3[2] * in[2] + W3[3] * in[3];
    float z1 = b3[1] + W3[4] * in[0] + W3[5] * in[1] + W3[6] * in[2] + W3[7] * in[3];

    int best = 0; float bd = 3.4e38f;
    #pragma unroll
    for (int k = 0; k < 3; ++k) {
        float dx = z0 - emb[2 * k], dy = z1 - emb[2 * k + 1];
        float d = dx * dx + dy * dy;
        if (d < bd) { bd = d; best = k; }   // strict < == argmin-first semantics
    }
    float q0 = emb[2 * best], q1 = emb[2 * best + 1];
    q[((size_t)b * 2 + 0) * 4096 + pix] = q0;
    q[((size_t)b * 2 + 1) * 4096 + pix] = q1;

    float d0 = q0 - z0, d1 = q1 - z1;
    ls[threadIdx.x] = d0 * d0 + d1 * d1;
    __syncthreads();
    for (int off = 128; off > 0; off >>= 1) {
        if ((int)threadIdx.x < off) ls[threadIdx.x] += ls[threadIdx.x + off];
        __syncthreads();
    }
    if (threadIdx.x == 0) lossPart[blockIdx.x] = ls[0];
}

// codebook + 0.2*commitment == 1.2 * mean((q-z)^2) in forward value.
__global__ void loss_final_kernel(const float* __restrict__ lossPart,
                                  float* __restrict__ out_loss, int nparts) {
    __shared__ float ls[256];
    float s = 0.f;
    for (int i = threadIdx.x; i < nparts; i += 256) s += lossPart[i];
    ls[threadIdx.x] = s;
    __syncthreads();
    for (int off = 128; off > 0; off >>= 1) {
        if ((int)threadIdx.x < off) ls[threadIdx.x] += ls[threadIdx.x + off];
        __syncthreads();
    }
    if (threadIdx.x == 0) out_loss[0] = 1.2f * ls[0] / 1048576.f;  // N*C
}

// ---------------------------------------------------------------------------
// deconv1: q[128,2,64,64] -> y3 raw [128,16,128,128], ConvTranspose k4 s2 p1.
// K = cin(2) x 4 x 4 = exactly 32; parity-invalid taps carry 0 in B.
// Wt1 torch layout [cin=2][cout=16][r][s]. Epilogue: stats partials (32/blk).
// ---------------------------------------------------------------------------
__global__ void deconv1_wmma_kernel(const float* __restrict__ q,
                                    const float* __restrict__ Wt1,
                                    const float* __restrict__ bt1,
                                    float* __restrict__ y3,
                                    float* __restrict__ part) {
    __shared__ float wred[8][32];
    const int lane = threadIdx.x & 31;
    const int wave = threadIdx.x >> 5;
    const int tile = blockIdx.x * 8 + wave;   // 131072 tiles
    const int b    = tile >> 10;
    const int rem  = tile & 1023;
    const int oy   = rem >> 3;
    const int ox0  = (rem & 7) << 4;

    const int m   = lane & 15;
    const int kbA = (lane < 16) ? 0 : 8;
    const int kbB = (lane < 16) ? 0 : 16;
    const int ox  = ox0 + (lane & 15);

    v16h a, bm;
    #pragma unroll
    for (int e = 0; e < 16; ++e) {
        int kg  = kbB + e;  // kg = cin*16 + r*4 + s
        int cin = kg >> 4, rs = kg & 15;
        a[e] = (_Float16)Wt1[cin * 256 + m * 16 + rs];
    }
    #pragma unroll
    for (int e = 0; e < 16; ++e) {
        int kg  = kbB + e;
        int cin = kg >> 4;
        int r   = (kg >> 2) & 3, s = kg & 3;
        int ty  = oy + 1 - r, tx = ox + 1 - s;          // iy = (oy + pad - r)/stride
        float v = 0.f;
        if (ty >= 0 && !(ty & 1) && tx >= 0 && !(tx & 1)) {
            int iy = ty >> 1, ix = tx >> 1;
            if (iy < 64 && ix < 64)
                v = q[(((size_t)b * 2 + cin) * 64 + iy) * 64 + ix];
        }
        bm[e] = (_Float16)v;
    }
    v8f c = {};
    c = __builtin_amdgcn_wmma_f32_16x16x32_f16(false, a, false, bm, (short)0, c, false, false);

    const int mb = (lane < 16) ? 0 : 8;
    float vals[8];
    #pragma unroll
    for (int r = 0; r < 8; ++r) {
        int mm = mb + r;
        float v = c[r] + bt1[mm];
        vals[r] = v;
        y3[(((size_t)b * 16 + mm) * 128 + oy) * 128 + ox] = v;
    }

    float sums[8], sqs[8];
    #pragma unroll
    for (int r = 0; r < 8; ++r) { sums[r] = vals[r]; sqs[r] = vals[r] * vals[r]; }
    #pragma unroll
    for (int mask = 1; mask < 16; mask <<= 1) {
        #pragma unroll
        for (int r = 0; r < 8; ++r) {
            sums[r] += __shfl_xor(sums[r], mask, 32);
            sqs[r]  += __shfl_xor(sqs[r],  mask, 32);
        }
    }
    if ((lane & 15) == 0) {
        #pragma unroll
        for (int r = 0; r < 8; ++r) {
            wred[wave][mb + r]      = sums[r];
            wred[wave][16 + mb + r] = sqs[r];
        }
    }
    __syncthreads();
    if (threadIdx.x < 32) {
        float t = 0.f;
        #pragma unroll
        for (int w = 0; w < 8; ++w) t += wred[w][threadIdx.x];
        part[(size_t)blockIdx.x * 32 + threadIdx.x] = t;
    }
}

// ---------------------------------------------------------------------------
// deconv2 + tanh: y3 raw -> out[128,1,256,256]. Single output channel =>
// rank-1 in WMMA terms; VALU dot with BN(layer3)+ReLU applied per load.
// Wt2 torch layout [cin=16][cout=1][r][s].
// ---------------------------------------------------------------------------
__global__ void deconv2_tanh_kernel(const float* __restrict__ y3,
                                    const float* __restrict__ scale3,
                                    const float* __restrict__ shift3,
                                    const float* __restrict__ Wt2,
                                    const float* __restrict__ bt2,
                                    float* __restrict__ out) {
    const unsigned long long p = (unsigned long long)blockIdx.x * 256 + threadIdx.x;
    const int b   = (int)(p >> 16);
    const int pix = (int)(p & 65535ull);
    const int oy  = pix >> 8;
    const int ox  = pix & 255;

    float acc = bt2[0];
    #pragma unroll
    for (int r = 0; r < 4; ++r) {
        int ty = oy + 1 - r;
        if (ty < 0 || (ty & 1)) continue;
        int iy = ty >> 1;
        if (iy >= 128) continue;
        #pragma unroll
        for (int s = 0; s < 4; ++s) {
            int tx = ox + 1 - s;
            if (tx < 0 || (tx & 1)) continue;
            int ix = tx >> 1;
            if (ix >= 128) continue;
            #pragma unroll
            for (int cin = 0; cin < 16; ++cin) {
                float v = y3[(((size_t)b * 16 + cin) * 128 + iy) * 128 + ix];
                v = v * scale3[cin] + shift3[cin];
                v = v > 0.f ? v : 0.f;
                acc += Wt2[cin * 16 + r * 4 + s] * v;
            }
        }
    }
    out[p] = tanhf(acc);
}

// ---------------------------------------------------------------------------
extern "C" void kernel_launch(void* const* d_in, const int* in_sizes, int n_in,
                              void* d_out, int out_size, void* d_ws, size_t ws_size,
                              hipStream_t stream) {
    (void)in_sizes; (void)n_in; (void)out_size; (void)ws_size;

    const float* x   = (const float*)d_in[0];
    const float* W1  = (const float*)d_in[1];
    const float* b1  = (const float*)d_in[2];
    const float* g1  = (const float*)d_in[3];
    const float* be1 = (const float*)d_in[4];
    const float* W2  = (const float*)d_in[5];
    const float* b2  = (const float*)d_in[6];
    const float* g2  = (const float*)d_in[7];
    const float* be2 = (const float*)d_in[8];
    const float* W3  = (const float*)d_in[9];
    const float* b3  = (const float*)d_in[10];
    const float* emb = (const float*)d_in[11];
    const float* Wt1 = (const float*)d_in[12];
    const float* bt1 = (const float*)d_in[13];
    const float* g3  = (const float*)d_in[14];
    const float* be3 = (const float*)d_in[15];
    const float* Wt2 = (const float*)d_in[16];
    const float* bt2 = (const float*)d_in[17];

    float* out = (float*)d_out;                  // 8,388,608 image + 1 loss

    // Workspace layout (floats). y1 buffer reused as y3 in the decoder.
    float* ws   = (float*)d_ws;
    float* y1   = ws;                            // 33,554,432 (raw conv1 / deconv1 out)
    float* y2   = y1 + 33554432ull;              //  2,097,152 (raw conv2 out)
    float* qb   = y2 + 2097152ull;               //  1,048,576 (quantized latents)
    float* sc   = qb + 1048576ull;               // 72 BN fold params
    float* scale1 = sc,      *shift1 = sc + 16;
    float* scale2 = sc + 32, *shift2 = sc + 36;
    float* scale3 = sc + 40, *shift3 = sc + 56;
    float* part1 = sc + 72;                      // 16384*32 = 524,288 (reused layer3)
    float* part2 = part1 + 524288ull;            //  4096*8  =  32,768
    float* lossPart = part2 + 32768ull;          //  2,048

    // ---- encoder ----
    conv1_wmma_kernel<<<16384, 256, 0, stream>>>(x, W1, b1, y1, part1);
    stats_scale_kernel<<<16, 256, 0, stream>>>(part1, 16384, 16, g1, be1,
                                               1.f / 2097152.f, scale1, shift1);
    conv2_wmma_kernel<<<4096, 256, 0, stream>>>(y1, scale1, shift1, W2, b2, y2, part2);
    stats_scale_kernel<<<4, 256, 0, stream>>>(part2, 4096, 4, g2, be2,
                                              1.f / 524288.f, scale2, shift2);

    // ---- bottleneck: 1x1 conv + VQ + loss ----
    conv3_vq_kernel<<<2048, 256, 0, stream>>>(y2, scale2, shift2, W3, b3, emb, qb, lossPart);
    loss_final_kernel<<<1, 256, 0, stream>>>(lossPart, out + 8388608, 2048);

    // ---- decoder (y1 buffer reused as y3) ----
    deconv1_wmma_kernel<<<16384, 256, 0, stream>>>(qb, Wt1, bt1, y1, part1);
    stats_scale_kernel<<<16, 256, 0, stream>>>(part1, 16384, 16, g3, be3,
                                               1.f / 2097152.f, scale3, shift3);
    deconv2_tanh_kernel<<<32768, 256, 0, stream>>>(y1, scale3, shift3, Wt2, bt2, out);
}
